// cAttn_87892210746037
// MI455X (gfx1250) — matrix operand order repaired and checked by reading, not compiled, not run on hardware
//
#include <hip/hip_runtime.h>

typedef __bf16 bf16_t;
typedef __attribute__((ext_vector_type(16))) __bf16 v16bf;
typedef __attribute__((ext_vector_type(8)))  float  v8f;

__device__ __forceinline__ bf16_t f2bf(float f) {
  unsigned u = __builtin_bit_cast(unsigned, f);
  unsigned r = (u + 0x7FFFu + ((u >> 16) & 1u)) >> 16;
  unsigned short s = (unsigned short)r;
  return __builtin_bit_cast(bf16_t, s);
}

#define SWZ_XOR_ADD(v, x)                                                     \
  (v) += __builtin_bit_cast(float, __builtin_amdgcn_ds_swizzle(               \
            __builtin_bit_cast(int, (v)), (((x) << 10) | 0x1F)))

// ---------------------------------------------------------------------------
// fp32 -> bf16 elementwise (used for x and z)
// ---------------------------------------------------------------------------
__global__ __launch_bounds__(256) void cvt_kernel(const float* __restrict__ src,
                                                  bf16_t* __restrict__ dst, int n) {
  int i = blockIdx.x * blockDim.x + threadIdx.x;
  if (i < n) dst[i] = f2bf(src[i]);
}

// ---------------------------------------------------------------------------
// Weight transpose + convert: W[k][n] fp32 -> Wt[n][k] bf16, via LDS 32x32
// tile so both global streams are coalesced. blockDim (32,8), grid (32,32,4).
// ---------------------------------------------------------------------------
__global__ __launch_bounds__(256) void wtrans_kernel(
    const float* __restrict__ Wq, const float* __restrict__ Wk,
    const float* __restrict__ Wv, const float* __restrict__ Wo,
    bf16_t* __restrict__ Wqt, bf16_t* __restrict__ Wkt,
    bf16_t* __restrict__ Wvt, bf16_t* __restrict__ Wot) {
  __shared__ float tile[32][33];
  const float* W;
  bf16_t* Wt;
  switch (blockIdx.z) {
    case 0:  W = Wq; Wt = Wqt; break;
    case 1:  W = Wk; Wt = Wkt; break;
    case 2:  W = Wv; Wt = Wvt; break;
    default: W = Wo; Wt = Wot; break;
  }
  const int tx = threadIdx.x;          // 0..31
  const int ty = threadIdx.y;          // 0..7
  const int n0 = blockIdx.x * 32;
  const int k0 = blockIdx.y * 32;
#pragma unroll
  for (int i = 0; i < 4; ++i)
    tile[ty + i * 8][tx] = W[(size_t)(k0 + ty + i * 8) * 1024 + n0 + tx];
  __syncthreads();
#pragma unroll
  for (int i = 0; i < 4; ++i)
    Wt[(size_t)(n0 + ty + i * 8) * 1024 + k0 + tx] = f2bf(tile[tx][ty + i * 8]);
}

// ---------------------------------------------------------------------------
// 4096x1024x1024 GEMM, bf16 WMMA, fp32 accumulate. All operands pre-converted
// bf16; W pre-transposed [n][k] so each lane's B operand is one contiguous
// 32B global load. No LDS, no barriers. Each wave computes a 16(M) x 64(N)
// strip: one A tile feeds 4 independent WMMAs per 32-wide K step.
// mode 0/1/2: C = xb @ {Wq,Wk,Wv} + b, scattered to per-head layouts (bf16)
// mode 3:     C = Obuf @ Wo + bo, plain row-major fp32 to d_out
// ---------------------------------------------------------------------------
__global__ __launch_bounds__(256) void gemm_kernel(
    const bf16_t* __restrict__ A,      // [4096 x 1024] bf16 row-major
    const bf16_t* __restrict__ Wqt, const bf16_t* __restrict__ Wkt,
    const bf16_t* __restrict__ Wvt, const bf16_t* __restrict__ Wot,
    const float* __restrict__ bq, const float* __restrict__ bk,
    const float* __restrict__ bvv, const float* __restrict__ bo,
    bf16_t* __restrict__ Qb, bf16_t* __restrict__ Kb, bf16_t* __restrict__ Vt,
    float* __restrict__ outp, int mode0) {
  constexpr int Kdim = 1024, N = 1024, Sl = 1024, HID = 64, NH = 16;

  const int mode = mode0 + blockIdx.y;
  const bf16_t* Wt;
  const float* bias;
  switch (mode) {
    case 0:  Wt = Wqt; bias = bq;  break;
    case 1:  Wt = Wkt; bias = bk;  break;
    case 2:  Wt = Wvt; bias = bvv; break;
    default: Wt = Wot; bias = bo;  break;
  }

  const int nchunk = blockIdx.x & 15;  // 16 chunks of 64 cols
  const int mc     = blockIdx.x >> 4;  // 32 chunks of 128 rows
  const int wid    = threadIdx.x >> 5;
  const int lane   = threadIdx.x & 31;
  const int n0     = nchunk * 64;
  const int m0     = mc * 128 + wid * 16;

  const int row  = lane & 15;          // A row (M)
  const int half = lane >> 4;
  const int c0   = half * 16;          // A K-phase
  const int kb   = half * 16;          // B K-phase
  const int ncol = lane & 15;          // B/C column (N)

  v8f c[4] = {};

  for (int kk0 = 0; kk0 < Kdim; kk0 += 32) {
    v16bf a;
    const bf16_t* ap = A + (size_t)(m0 + row) * Kdim + kk0 + c0;
#pragma unroll
    for (int j = 0; j < 8; ++j) a[j] = ap[j];
#pragma unroll
    for (int j = 0; j < 8; ++j) a[8 + j] = ap[8 + j];

#pragma unroll
    for (int nt = 0; nt < 4; ++nt) {
      v16bf b =
          *(const v16bf*)(Wt + (size_t)(n0 + nt * 16 + ncol) * Kdim + kk0 + kb);
      c[nt] = __builtin_amdgcn_wmma_f32_16x16x32_bf16(false, a, false, b,
                                                      (short)0, c[nt], false,
                                                      false);
    }
  }

#pragma unroll
  for (int nt = 0; nt < 4; ++nt) {
    const int col = n0 + nt * 16 + ncol;
    const float bc = bias[col];
    if (mode == 3) {
#pragma unroll
      for (int r = 0; r < 8; ++r) {
        int m = m0 + r + half * 8;
        outp[(size_t)m * N + col] = c[nt][r] + bc;
      }
    } else {
      const int nh = col & (NH - 1);   // head index (innermost in reference)
      const int h  = col >> 4;         // hid index
#pragma unroll
      for (int r = 0; r < 8; ++r) {
        int m = m0 + r + half * 8;
        int b = m >> 10, s = m & 1023;
        float v = c[nt][r] + bc;
        if (mode == 0)
          Qb[(((size_t)(b * NH + nh)) * Sl + s) * HID + h] = f2bf(v);
        else if (mode == 1)
          Kb[(((size_t)(b * NH + nh)) * Sl + s) * HID + h] = f2bf(v);
        else
          Vt[(((size_t)(b * NH + nh)) * HID + h) * Sl + s] = f2bf(v);
      }
    }
  }
}

// ---------------------------------------------------------------------------
// Fused attention: per (b, head, 16-row q-tile) one wave streams all 1024
// keys in 32-wide chunks: score WMMAs, zz WMMA (KC=32 = one WMMA K),
// p = zz * exp(score/8) (masked rows -> p = zz), writes unnormalized attn,
// accumulates rowsum + p@V in registers. Head output normalized here and
// stored bf16; rowsums saved for the attn rescale pass.
// ---------------------------------------------------------------------------
__global__ __launch_bounds__(256) void attn_kernel(
    const bf16_t* __restrict__ Qb, const bf16_t* __restrict__ Kb,
    const bf16_t* __restrict__ Vt, const bf16_t* __restrict__ zb,
    const unsigned char* __restrict__ pmask, float* __restrict__ attn,
    bf16_t* __restrict__ Ob, float* __restrict__ rs) {
  constexpr int Sl = 1024, HID = 64, NH = 16;
  __shared__ alignas(32) bf16_t pbuf[8][16 * 32];  // per-wave p tile (q x k)

  const int tid  = threadIdx.x;
  const int wid  = tid >> 5;
  const int lane = tid & 31;
  const int qblk = blockIdx.x & 7;
  const int n    = (blockIdx.x >> 3) & 15;
  const int b    = blockIdx.x >> 7;
  const int q0   = qblk * 128 + wid * 16;

  const bf16_t* Qp = Qb + ((size_t)(b * NH + n)) * Sl * HID;
  const bf16_t* Kp = Kb + ((size_t)(b * NH + n)) * Sl * HID;
  const bf16_t* Vp = Vt + ((size_t)(b * NH + n)) * HID * Sl;
  const bf16_t* zp = zb + (size_t)b * Sl * 32;
  float* Ap        = attn + ((size_t)(b * NH + n)) * Sl * Sl;

  const int row  = lane & 15;
  const int half = lane >> 4;
  const int c0   = half * 16;
  const int kb   = half * 16;
  const int ncol = lane & 15;

  v16bf aq0, aq1, az;
  {
    const bf16_t* qp = Qp + (size_t)(q0 + row) * HID;
#pragma unroll
    for (int j = 0; j < 8; ++j) { aq0[j] = qp[c0 + j];      aq0[8 + j] = qp[c0 + 8 + j]; }
#pragma unroll
    for (int j = 0; j < 8; ++j) { aq1[j] = qp[32 + c0 + j]; aq1[8 + j] = qp[32 + c0 + 8 + j]; }
    const bf16_t* zq = zp + (size_t)(q0 + row) * 32;
#pragma unroll
    for (int j = 0; j < 8; ++j) { az[j] = zq[c0 + j];       az[8 + j] = zq[c0 + 8 + j]; }
  }

  bool msk[8];
#pragma unroll
  for (int r = 0; r < 8; ++r)
    msk[r] = pmask[(size_t)b * Sl + q0 + r + half * 8] != 0;

  float rsum[8];
#pragma unroll
  for (int r = 0; r < 8; ++r) rsum[r] = 0.0f;

  v8f oacc[4] = {};  // 16q x 64h fp32 accumulators

  for (int k0 = 0; k0 < Sl; k0 += 32) {
    if (k0 + 32 < Sl) {
      __builtin_prefetch(Kp + (size_t)(k0 + 32 + ncol) * HID, 0, 1);
      __builtin_prefetch(Vp + (size_t)ncol * Sl + k0 + 32, 0, 1);
    }
#pragma unroll
    for (int sub = 0; sub < 2; ++sub) {
      const int kk = k0 + sub * 16;
      v16bf bk0 = *(const v16bf*)(Kp + (size_t)(kk + ncol) * HID + kb);
      v16bf bk1 = *(const v16bf*)(Kp + (size_t)(kk + ncol) * HID + 32 + kb);
      v8f cs = {};
      cs = __builtin_amdgcn_wmma_f32_16x16x32_bf16(false, aq0, false, bk0,
                                                   (short)0, cs, false, false);
      cs = __builtin_amdgcn_wmma_f32_16x16x32_bf16(false, aq1, false, bk1,
                                                   (short)0, cs, false, false);
      v16bf bz = *(const v16bf*)(zp + (size_t)(kk + ncol) * 32 + kb);
      v8f cz = {};
      cz = __builtin_amdgcn_wmma_f32_16x16x32_bf16(false, az, false, bz,
                                                   (short)0, cz, false, false);
#pragma unroll
      for (int r = 0; r < 8; ++r) {
        // masked query rows: softmax of constant row -> p = zz exactly
        float p = msk[r] ? cz[r] : cz[r] * __expf(cs[r] * 0.125f);
        rsum[r] += p;
        Ap[(size_t)(q0 + r + half * 8) * Sl + kk + ncol] = p;  // unnormalized
        pbuf[wid][(r + half * 8) * 32 + sub * 16 + ncol] = f2bf(p);
      }
    }
    // redistribute p (C layout) -> A layout via wave-private LDS slice
    v16bf ap;
    {
      const bf16_t* pp = &pbuf[wid][row * 32];
#pragma unroll
      for (int j = 0; j < 8; ++j) { ap[j] = pp[c0 + j]; ap[8 + j] = pp[c0 + 8 + j]; }
    }
#pragma unroll
    for (int ht = 0; ht < 4; ++ht) {
      v16bf bv = *(const v16bf*)(Vp + (size_t)(ht * 16 + ncol) * Sl + k0 + kb);
      oacc[ht] = __builtin_amdgcn_wmma_f32_16x16x32_bf16(false, ap, false, bv,
                                                         (short)0, oacc[ht],
                                                         false, false);
    }
  }

  // rowsum: xor-reduce over the 16 lanes of each half (columns of C)
#pragma unroll
  for (int r = 0; r < 8; ++r) {
    float v = rsum[r];
    SWZ_XOR_ADD(v, 1);
    SWZ_XOR_ADD(v, 2);
    SWZ_XOR_ADD(v, 4);
    SWZ_XOR_ADD(v, 8);
    rsum[r] = v;
  }

  float inv[8];
#pragma unroll
  for (int r = 0; r < 8; ++r) inv[r] = 1.0f / rsum[r];

  // normalized head output, merged layout [b, s, n*HID + h] (head outermost)
#pragma unroll
  for (int ht = 0; ht < 4; ++ht) {
#pragma unroll
    for (int r = 0; r < 8; ++r) {
      int q = q0 + r + half * 8;
      Ob[((size_t)(b * Sl + q)) * 1024 + n * HID + ht * 16 + ncol] =
          f2bf(oacc[ht][r] * inv[r]);
    }
  }
  if (ncol == 0) {
#pragma unroll
    for (int r = 0; r < 8; ++r)
      rs[(size_t)(b * NH + n) * Sl + q0 + r + half * 8] = rsum[r];
  }
}

// ---------------------------------------------------------------------------
// attn row rescale: attn[row,:] *= 1/rowsum. One block per row, float4 lanes.
// ---------------------------------------------------------------------------
__global__ __launch_bounds__(256) void rescale_kernel(float* __restrict__ attn,
                                                      const float* __restrict__ rs) {
  const size_t rowid = blockIdx.x;
  const float inv = 1.0f / rs[rowid];
  float4* p = (float4*)(attn + rowid * 1024);
  float4 v = p[threadIdx.x];
  v.x *= inv; v.y *= inv; v.z *= inv; v.w *= inv;
  p[threadIdx.x] = v;
}

extern "C" void kernel_launch(void* const* d_in, const int* in_sizes, int n_in,
                              void* d_out, int out_size, void* d_ws, size_t ws_size,
                              hipStream_t stream) {
  const float* x          = (const float*)d_in[0];
  const float* z          = (const float*)d_in[1];
  const unsigned char* pm = (const unsigned char*)d_in[2];  // jax bool = 1 byte
  const float* Wq = (const float*)d_in[3];
  const float* Wk = (const float*)d_in[4];
  const float* Wv = (const float*)d_in[5];
  const float* Wo = (const float*)d_in[6];
  const float* bq = (const float*)d_in[7];
  const float* bk = (const float*)d_in[8];
  const float* bv = (const float*)d_in[9];
  const float* bo = (const float*)d_in[10];

  float* attn = (float*)d_out;                       // [4,16,1024,1024] fp32
  float* outp = attn + (size_t)4 * 16 * 1024 * 1024; // [4,1024,1024] fp32

  char* ws = (char*)d_ws;
  bf16_t* Qb  = (bf16_t*)ws; ws += (size_t)4 * 16 * 1024 * 64 * 2;  // 8 MB
  bf16_t* Kb  = (bf16_t*)ws; ws += (size_t)4 * 16 * 1024 * 64 * 2;  // 8 MB
  bf16_t* Vt  = (bf16_t*)ws; ws += (size_t)4 * 16 * 1024 * 64 * 2;  // 8 MB
  bf16_t* Ob  = (bf16_t*)ws; ws += (size_t)4 * 1024 * 1024 * 2;     // 8 MB
  bf16_t* xb  = (bf16_t*)ws; ws += (size_t)4 * 1024 * 1024 * 2;     // 8 MB
  bf16_t* zb  = (bf16_t*)ws; ws += (size_t)4 * 1024 * 32 * 2;       // 256 KB
  float*  rsb = (float*)ws;  ws += (size_t)4 * 16 * 1024 * 4;       // 256 KB
  bf16_t* Wqt = (bf16_t*)ws; ws += (size_t)1024 * 1024 * 2;         // 2 MB
  bf16_t* Wkt = (bf16_t*)ws; ws += (size_t)1024 * 1024 * 2;         // 2 MB
  bf16_t* Wvt = (bf16_t*)ws; ws += (size_t)1024 * 1024 * 2;         // 2 MB
  bf16_t* Wot = (bf16_t*)ws; ws += (size_t)1024 * 1024 * 2;         // 2 MB

  cvt_kernel<<<dim3(16384), dim3(256), 0, stream>>>(x, xb, 4 * 1024 * 1024);
  cvt_kernel<<<dim3(512), dim3(256), 0, stream>>>(z, zb, 4 * 1024 * 32);
  wtrans_kernel<<<dim3(32, 32, 4), dim3(32, 8), 0, stream>>>(
      Wq, Wk, Wv, Wo, Wqt, Wkt, Wvt, Wot);

  gemm_kernel<<<dim3(512, 3), dim3(256), 0, stream>>>(
      xb, Wqt, Wkt, Wvt, Wot, bq, bk, bv, bo, Qb, Kb, Vt, outp, 0);

  attn_kernel<<<dim3(512), dim3(256), 0, stream>>>(Qb, Kb, Vt, zb, pm, attn,
                                                   Ob, rsb);

  gemm_kernel<<<dim3(512, 1), dim3(256), 0, stream>>>(
      Ob, Wqt, Wkt, Wvt, Wot, bq, bk, bv, bo, Qb, Kb, Vt, outp, 3);

  rescale_kernel<<<dim3(4 * 16 * 1024), dim3(256), 0, stream>>>(attn, rsb);
}
